// TreeModel_44032004719312
// MI455X (gfx1250) — matrix operand order repaired and checked by reading, not compiled, not run on hardware
//
#include <hip/hip_runtime.h>

typedef __attribute__((ext_vector_type(16))) __bf16 v16bf;
typedef __attribute__((ext_vector_type(8)))  __bf16 v8bf;
typedef __attribute__((ext_vector_type(8)))  float  v8f;

#define QD     64
#define VD     64
#define BT     16     // queries per block (WMMA M)
#define TOPD   8      // top levels handled by scalar path walk
#define SUBD   8      // subtree depth handled by WMMA expansion
#define NLEAF  256    // leaves per subtree
#define VPITCH 272    // halves per row of transposed V stage (32B-aligned rows, bank-skewed)

static __device__ __forceinline__ __bf16 f2bf(float f) {
  unsigned u = __builtin_bit_cast(unsigned, f);
  u += 0x7FFFu + ((u >> 16) & 1u);             // round-to-nearest-even
  unsigned short h = (unsigned short)(u >> 16);
  return __builtin_bit_cast(__bf16, h);
}

// A-fragment (16-bit, 16x32): lane half-wave holds K-chunks [hi,hi+8) and [hi+16,hi+24)
static __device__ __forceinline__ v16bf afrag_chunks(const __bf16* p) {
  v8bf a = *(const v8bf*)p;
  v8bf b = *(const v8bf*)(p + 16);
  return __builtin_shufflevector(a, b, 0,1,2,3,4,5,6,7,8,9,10,11,12,13,14,15);
}

static __device__ __forceinline__ v8f wmma_bf16(v16bf a, v16bf b, v8f c) {
  return __builtin_amdgcn_wmma_f32_16x16x32_bf16(false, a, false, b, (short)0, c, false, false);
}

// sigmoid via hardware reciprocal (v_rcp_f32, ~1 ulp) instead of IEEE division
static __device__ __forceinline__ float fast_sigmoid(float z) {
  return __builtin_amdgcn_rcpf(1.f + __expf(-z));
}

// log(sigmoid(z)), numerically stable
static __device__ __forceinline__ float log_sigmoid(float z) {
  return (z > 0.f) ? -__logf(1.f + __expf(-z)) : z - __logf(1.f + __expf(z));
}

__global__ __launch_bounds__(256)
void tree_attn_kernel(const float* __restrict__ Q,
                      const float* __restrict__ TK,
                      const float* __restrict__ TV,
                      float* __restrict__ OUT)
{
  __shared__ __align__(32) unsigned char smem[57856];
  float*  qf    = (float*)(smem);            // 16*64*4  = 4096
  __bf16* qbf   = (__bf16*)(smem + 4096);    // 16*64*2  = 2048
  float*  logp  = (float*)(smem + 6144);     // 16*8*4   = 512
  float*  probs = (float*)(smem + 6656);     // 16*256*4 = 16384 (in-place expanded)
  __bf16* kv    = (__bf16*)(smem + 23040);   // 34816: key-diffs [node][64] bf16 / V [vd][272] bf16

  const int tid   = threadIdx.x;
  const int lane  = tid & 31;
  const int wave  = tid >> 5;
  const int s     = blockIdx.x;              // subtree id at level TOPD, [0,256)
  const int qbase = blockIdx.y * BT;
  const int n16   = lane & 15;
  const int hiK   = (lane >= 16) ? 8  : 0;   // C-frag M offset
  const int hiB   = (lane >= 16) ? 16 : 0;   // B-frag K offset (contiguous-K layout)

  // ---- 1) stage Q tile (f32 for path walk, bf16 for WMMA) ----
  for (int i = tid; i < BT * QD; i += 256) {
    float v = Q[(size_t)(qbase + (i >> 6)) * QD + (i & 63)];
    qf[i] = v; qbf[i] = f2bf(v);
  }
  __syncthreads();

  // ---- 2) top-path log-prob of this subtree's root, per query ----
  // softmax over 2 children == sigmoid of the logit difference z = q.(k0-k1)
  if (tid < BT * TOPD) {
    int qi = tid >> 3, d = tid & 7;
    int anc  = s >> (TOPD - d);
    int cd   = (s >> (TOPD - 1 - d)) & 1;    // which child the path takes
    int node = (1 << d) - 1 + anc;
    const float* k0 = TK + (size_t)node * 2 * QD;
    const float* k1 = k0 + QD;
    const float* qv = qf + qi * QD;
    float z = 0.f;
    #pragma unroll 8
    for (int i = 0; i < QD; ++i) z += qv[i] * (k0[i] - k1[i]);
    logp[qi * 8 + d] = log_sigmoid(cd ? -z : z);
  }
  __syncthreads();
  if (tid < BT) {
    float sum = 0.f;
    #pragma unroll
    for (int d = 0; d < 8; ++d) sum += logp[tid * 8 + d];
    probs[tid * NLEAF] = __expf(sum);        // prefix prob seeds in-place expansion
  }
  __syncthreads();

  // ---- Q A-fragments (level-invariant) ----
  v16bf aq0 = *(const v16bf*)(qbf + n16 * QD + 0  + hiB);
  v16bf aq1 = *(const v16bf*)(qbf + n16 * QD + 32 + hiB);

  // ---- 3) subtree expansion, levels e = 0..7 ----
  for (int e = 0; e < SUBD; ++e) {
    const int nNodes   = 1 << e;                              // one key-diff column per node
    const int baseNode = (1 << (TOPD + e)) - 1 + s * nNodes;

    // stage bf16 key differences: kv[node][k] = k_left - k_right
    for (int i = tid; i < nNodes * QD; i += 256) {
      int j = i >> 6, k = i & 63;
      const float* kp = TK + ((size_t)(baseNode + j) * 2) * QD + k;
      kv[j * QD + k] = f2bf(kp[0] - kp[QD]);
    }

    // prefetch next level's keys (or the V tile) into cache while we compute
    if (e + 1 < SUBD) {
      int b2    = (1 << (TOPD + e + 1)) - 1 + s * (1 << (e + 1));
      int lines = (1 << (e + 1)) * 2;                         // 256B lines covering 2 keys/node
      for (int i = tid; i < lines; i += 256)
        __builtin_prefetch(TK + (size_t)b2 * 2 * QD + (size_t)i * 64, 0, 3);
    } else {
      for (int i = tid; i < NLEAF; i += 256)                  // one 256B line per leaf row
        __builtin_prefetch(TV + ((size_t)s * NLEAF + i) * VD, 0, 3);
    }
    __syncthreads();

    // one 16-node tile per wave, wave-uniform branch (EXEC all-1s inside for WMMA)
    const int T = (nNodes + 15) >> 4;                         // tiles this level (max 8 at e=7)
    const int j = wave * 16 + n16;                            // node column owned by this lane
    float pv[8], ov[8];
    if (wave < T) {
      v16bf b0 = *(const v16bf*)(kv + j * QD + hiB);          // B: contiguous 16 K per half-wave
      v16bf b1 = *(const v16bf*)(kv + j * QD + 32 + hiB);
      v8f c = {0.f,0.f,0.f,0.f,0.f,0.f,0.f,0.f};
      c = wmma_bf16(aq0, b0, c);
      c = wmma_bf16(aq1, b1, c);
      #pragma unroll
      for (int r = 0; r < 8; ++r) {                           // C: M = r + hiK, N = n16
        pv[r] = fast_sigmoid(c[r]);                           // p(left child)
        ov[r] = probs[(r + hiK) * NLEAF + j];                 // read phase (pre-barrier)
      }
    }
    __syncthreads();                                          // all reads before any writes
    if (wave < T && j < nNodes) {
      #pragma unroll
      for (int r = 0; r < 8; ++r) {
        float p0 = pv[r], op = ov[r];
        int base = (r + hiK) * NLEAF + 2 * j;
        probs[base]     = op * p0;
        probs[base + 1] = op * (1.f - p0);
      }
    }
    __syncthreads();
  }

  // ---- 4) stage V transposed: kv[vd][leaf], bf16, skewed pitch ----
  for (int i = tid; i < NLEAF * VD; i += 256) {
    int leaf = i >> 6, vd = i & 63;
    kv[vd * VPITCH + leaf] = f2bf(TV[((size_t)s * NLEAF + leaf) * VD + vd]);
  }
  __syncthreads();

  // ---- 5) out_tile(16x64) = P(16x256) @ V(256x64); split K across wave pairs ----
  {
    const int nt    = wave & 3;          // N-tile of 16 value dims
    const int kbase = (wave >> 2) * 128; // waves 0-3: K 0..127, waves 4-7: K 128..255
    v8f c = {0.f,0.f,0.f,0.f,0.f,0.f,0.f,0.f};
    #pragma unroll
    for (int kc = 0; kc < 4; ++kc) {
      int kk = kbase + kc * 32;
      v16bf a;
      #pragma unroll
      for (int h = 0; h < 16; ++h) {     // A contiguous-K: K = h + hiB
        int k = kk + h + hiB;
        a[h] = f2bf(probs[n16 * NLEAF + k]);
      }
      v16bf b = *(const v16bf*)(kv + (nt * 16 + n16) * VPITCH + kk + hiB);
      c = wmma_bf16(a, b, c);
    }
    #pragma unroll
    for (int r = 0; r < 8; ++r) {
      int row = qbase + r + hiK;
      int cv  = nt * 16 + n16;
      unsafeAtomicAdd(&OUT[(size_t)row * VD + cv], c[r]);     // 256 subtrees accumulate
    }
  }
}

extern "C" void kernel_launch(void* const* d_in, const int* in_sizes, int n_in,
                              void* d_out, int out_size, void* d_ws, size_t ws_size,
                              hipStream_t stream) {
  const float* q  = (const float*)d_in[0];  // (2048, 64)
  const float* tk = (const float*)d_in[1];  // (65535, 2, 64)
  const float* tv = (const float*)d_in[2];  // (65536, 64)
  float* out = (float*)d_out;               // (2048, 64)

  hipMemsetAsync(d_out, 0, (size_t)out_size * sizeof(float), stream);

  dim3 grid(256 /*subtrees*/, 2048 / BT /*batch tiles*/);
  dim3 block(256);
  tree_attn_kernel<<<grid, block, 0, stream>>>(q, tk, tv, out);
}
